// FNO_24438363915049
// MI455X (gfx1250) — matrix operand order repaired and checked by reading, not compiled, not run on hardware
//
#include <hip/hip_runtime.h>
#include <math.h>

typedef __attribute__((ext_vector_type(16))) __bf16          v16bf;
typedef __attribute__((ext_vector_type(16))) unsigned short  v16u;
typedef __attribute__((ext_vector_type(8)))  float           v8f;

#define HP 264              // padded spatial size
#define NPIX (264*264)      // 69696
#define NTILE (NPIX/16)     // 4356 pixel tiles per image
#define CW 64
#define BATCH 8
#define MODES 32
#define NROWS 64            // 2*MODES rows kept along H

__device__ __forceinline__ float gelu_exact(float v) {
    return 0.5f * v * (1.0f + erff(v * 0.70710678118654752f));
}

// bf16 h copy, pre-swizzled into WMMA B-fragment order:
// [b][ptile][chunk(2)][lane(32)][e(16)] ; lane = kh*16 + (p&15), i = ch*32 + kh*16 + e
__device__ __forceinline__ long hbf_index(long b, int i, long p) {
    long ptile = p >> 4;
    int  col = (int)(p & 15);
    int  ch = i >> 5;
    int  kl = i & 31;
    int  kh = kl >> 4;
    int  e  = kl & 15;
    return (((b * NTILE + ptile) * 2 + ch) * 32 + (kh * 16 + col)) * 16 + e;
}

// ---------------------------------------------------------------- twiddles
// tab1: [2][32][264]  cos/sin(2*pi*k*y/264), k<32   (W-direction DFT)
// tab2: [2][64][264]  cos/sin(2*pi*kx_r*h/264), kx_r = r<32? r : r+200
__global__ void fno_init_tabs(float* __restrict__ tab1, float* __restrict__ tab2) {
    int i = blockIdx.x * blockDim.x + threadIdx.x;
    const float w = 6.283185307179586f / 264.0f;
    if (i < 32 * HP) {
        int k = i / HP, y = i % HP;
        float a = w * (float)((k * y) % HP);
        tab1[i] = cosf(a);
        tab1[32 * HP + i] = sinf(a);
    }
    if (i < 64 * HP) {
        int r = i / HP, hh = i % HP;
        int kx = (r < 32) ? r : (r + 200);     // H-M + (r-32) = 232 + r - 32
        float a = w * (float)(((long)kx * hh) % HP);
        tab2[i] = cosf(a);
        tab2[64 * HP + i] = sinf(a);
    }
}

// ---------------------------------------------------------------- lift + pad
// x[8,3,256,256] -> h[8,64,264,264] (f32)  +  hbf (bf16, B-frag swizzled)
__global__ void fno_lift_pad(const float* __restrict__ x,
                             const float* __restrict__ w1, const float* __restrict__ b1,
                             const float* __restrict__ w2, const float* __restrict__ b2,
                             float* __restrict__ hbuf, __bf16* __restrict__ hbf) {
    long idx = (long)blockIdx.x * blockDim.x + threadIdx.x;   // over 8*264*264
    if (idx >= (long)BATCH * NPIX) return;
    int yy = (int)(idx % HP);
    long t  = idx / HP;
    int xx = (int)(t % HP);
    long b  = t / HP;
    long p  = (long)xx * HP + yy;
    float* dst = hbuf + b * (long)CW * NPIX + p;
    if (xx >= 256 || yy >= 256) {
        #pragma unroll 4
        for (int o = 0; o < CW; ++o) {
            dst[(long)o * NPIX] = 0.0f;
            hbf[hbf_index(b, o, p)] = (__bf16)0.0f;
        }
        return;
    }
    float f[5];
    #pragma unroll
    for (int c = 0; c < 3; ++c)
        f[c] = x[((b * 3 + c) * 256 + xx) * 256 + yy];
    f[3] = (float)xx * (1.0f / 255.0f);
    f[4] = (float)yy * (1.0f / 255.0f);
    float mid[32];
    #pragma unroll 4
    for (int j = 0; j < 32; ++j) {
        float s = b1[j];
        #pragma unroll
        for (int c = 0; c < 5; ++c) s += w1[j * 5 + c] * f[c];
        mid[j] = gelu_exact(s);
    }
    #pragma unroll 2
    for (int o = 0; o < CW; ++o) {
        float s = b2[o];
        #pragma unroll 4
        for (int j = 0; j < 32; ++j) s += w2[o * 32 + j] * mid[j];
        dst[(long)o * NPIX] = s;
        hbf[hbf_index(b, o, p)] = (__bf16)s;
    }
}

// ---------------------------------------------------------------- spectral weight staging (per layer)
// wcb2: bf16, A-fragment swizzled: [mode][reim(2)][ctile(4)][chunk(2)][lane(32)][e(16)]
//   A[o][i] = Wc[i][o];  o = t*16 + (lane&15),  i = ch*32 + klocal(e, lane>>4)
__global__ void fno_prep_weights(const float* __restrict__ spw1,
                                 const float* __restrict__ spw2,
                                 __bf16* __restrict__ wcb2, int layer) {
    long idx = (long)blockIdx.x * blockDim.x + threadIdx.x;   // 2048*2*4*2*32*16 = 16,777,216
    if (idx >= (long)NROWS * MODES * 2 * 4 * 2 * 32 * 16) return;
    int e    = (int)(idx & 15);
    int lane = (int)((idx >> 4) & 31);
    int ch   = (int)((idx >> 9) & 1);
    int t    = (int)((idx >> 10) & 3);
    int c    = (int)((idx >> 12) & 1);
    int mode = (int)(idx >> 13);               // r*32 + m
    int r = mode >> 5, m = mode & 31;
    int kh = lane >> 4;
    int kl = kh * 16 + e;
    int i  = ch * 32 + kl;
    int o  = t * 16 + (lane & 15);
    const float* src = (r < 32) ? spw1 : spw2;
    int mx = r & 31;
    long sidx = (((((long)layer * CW + i) * CW + o) * MODES + mx) * MODES + m) * 2 + c;
    wcb2[idx] = (__bf16)src[sidx];
}

// ---------------------------------------------------------------- 1x1 conv (WMMA bf16)
// y[b,o,p] = sum_i W[o,i] h[b,i,p] + bias[o]
// 8 waves/block; weights staged to LDS in A-frag order; B-frags from swizzled hbf.
__global__ void __launch_bounds__(256)
fno_conv1x1_wmma(const __bf16* __restrict__ hbf, const float* __restrict__ convw,
                 const float* __restrict__ bias, float* __restrict__ y) {
    __shared__ __attribute__((aligned(32))) __bf16 lw[4096];   // [t][ch][lane][e]
    for (int j = threadIdx.x; j < 4096; j += 256) {
        int e    = j & 15;
        int lane = (j >> 4) & 31;
        int ch   = (j >> 9) & 1;
        int t    = j >> 10;
        int o  = t * 16 + (lane & 15);
        int kh = lane >> 4;
        int kl = kh * 16 + e;
        int i  = ch * 32 + kl;
        lw[j] = (__bf16)convw[o * CW + i];
    }
    __syncthreads();

    const int lane = threadIdx.x & 31;
    const int wave = threadIdx.x >> 5;
    const long b = blockIdx.y;
    const long ptile = (long)blockIdx.x * 8 + wave;
    if (ptile >= NTILE) return;                 // wave-uniform exit (after barrier)

    const __bf16* hb = hbf + ((b * NTILE + ptile) * 2) * 512;
    v8f acc[4] = {v8f{}, v8f{}, v8f{}, v8f{}};
    #pragma unroll
    for (int ch = 0; ch < 2; ++ch) {
        v16bf bfrag = *(const v16bf*)(hb + ch * 512 + lane * 16);   // 32B contiguous/lane
        #pragma unroll
        for (int t = 0; t < 4; ++t) {
            v16bf afrag = *(const v16bf*)(&lw[((t * 2 + ch) * 32 + lane) * 16]);
            acc[t] = __builtin_amdgcn_wmma_f32_16x16x32_bf16(
                false, afrag, false, bfrag, (short)0, acc[t], false, false);
        }
    }
    const int col = lane & 15, kh = lane >> 4;
    const long p = ptile * 16 + col;
    #pragma unroll
    for (int t = 0; t < 4; ++t)
        #pragma unroll
        for (int v = 0; v < 8; ++v) {
            int o = t * 16 + v + 8 * kh;
            y[(b * CW + o) * (long)NPIX + p] = acc[t][v] + bias[o];
        }
}

// ---------------------------------------------------------------- forward DFT along W
// per (b,c,h) row: Xw[m] = sum_y h[y] * e^{-2pi i m y/264}, m<32
__global__ void fno_dft_w_fwd(const float* __restrict__ h, const float* __restrict__ tab1,
                              float* __restrict__ xwr, float* __restrict__ xwi) {
    __shared__ float row[HP];
    long rid = blockIdx.x;                      // 8*64*264 rows
    const float* src = h + rid * HP;
    for (int i = threadIdx.x; i < HP; i += 64) row[i] = src[i];
    __syncthreads();
    int m = threadIdx.x & 31;
    int part = threadIdx.x >> 5;
    float s = 0.0f;
    if (part == 0) {
        const float* tc = tab1 + m * HP;
        for (int y = 0; y < HP; ++y) s += row[y] * tc[y];
        xwr[rid * 32 + m] = s;
    } else {
        const float* ts = tab1 + 32 * HP + m * HP;
        for (int y = 0; y < HP; ++y) s -= row[y] * ts[y];
        xwi[rid * 32 + m] = s;
    }
}

// ---------------------------------------------------------------- forward DFT along H
// Xf[bc, r, m] = sum_h Xw[bc, h, m] * e^{-2pi i kx_r h/264}; stored bf16 interleaved
__global__ void fno_dft_h_fwd(const float* __restrict__ xwr, const float* __restrict__ xwi,
                              const float* __restrict__ tab2, __bf16* __restrict__ xfb) {
    long idx = (long)blockIdx.x * blockDim.x + threadIdx.x;   // 8*64*64*32
    if (idx >= (long)BATCH * CW * NROWS * MODES) return;
    int m = (int)(idx & 31);
    int r = (int)((idx >> 5) & 63);
    long bc = idx >> 11;
    const float* tc = tab2 + r * HP;
    const float* ts = tab2 + 64 * HP + r * HP;
    const float* xr = xwr + bc * (long)HP * 32 + m;
    const float* xi = xwi + bc * (long)HP * 32 + m;
    float sr = 0.0f, si = 0.0f;
    for (int hh = 0; hh < HP; ++hh) {
        float c = tc[hh], s = ts[hh];
        float a = xr[hh * 32], b = xi[hh * 32];
        sr += a * c + b * s;          // (a+ib)(c-is)
        si += b * c - a * s;
    }
    long o = (bc * NROWS + r) * MODES + m;
    xfb[o * 2]     = (__bf16)sr;
    xfb[o * 2 + 1] = (__bf16)si;
}

// ---------------------------------------------------------------- per-mode complex channel mix (WMMA)
// Yf[b,o] = sum_i Xf[b,i] * Wc[i,o] per mode; A = Wc^T (o x i), B = Xf (i x b)
__global__ void __launch_bounds__(32)
fno_specmix_wmma(const __bf16* __restrict__ xfb, const __bf16* __restrict__ wcb2,
                 float* __restrict__ yfr, float* __restrict__ yfi) {
    const int lane = threadIdx.x;
    const int col  = lane & 15;
    const int kh   = lane >> 4;
    const int mode = blockIdx.x;               // r*32 + m
    const int r = mode >> 5, m = mode & 31;
    const int t = blockIdx.y;                  // cout tile
    const __bf16* wbase = wcb2 + (long)mode * 8192;

    v8f accr = v8f{}, acci = v8f{};
    #pragma unroll
    for (int ch = 0; ch < 2; ++ch) {
        v16bf br, bi;
        #pragma unroll
        for (int e = 0; e < 16; ++e) {
            int i = ch * 32 + kh * 16 + e;
            if (col < BATCH) {
                long base = ((((long)col * CW + i) * NROWS + r) * MODES + m) * 2;
                br[e] = xfb[base];
                bi[e] = xfb[base + 1];
            } else {
                br[e] = (__bf16)0.0f;
                bi[e] = (__bf16)0.0f;
            }
        }
        v16bf ar = *(const v16bf*)(wbase + ((0 * 4 + t) * 2 + ch) * 512 + lane * 16);
        v16bf ai = *(const v16bf*)(wbase + ((1 * 4 + t) * 2 + ch) * 512 + lane * 16);
        v16u  aiu = __builtin_bit_cast(v16u, ai);
        aiu = aiu ^ (unsigned short)0x8000;                     // bf16 sign flip
        v16bf ain = __builtin_bit_cast(v16bf, aiu);
        accr = __builtin_amdgcn_wmma_f32_16x16x32_bf16(false, ar,  false, br, (short)0, accr, false, false);
        accr = __builtin_amdgcn_wmma_f32_16x16x32_bf16(false, ain, false, bi, (short)0, accr, false, false);
        acci = __builtin_amdgcn_wmma_f32_16x16x32_bf16(false, ar,  false, bi, (short)0, acci, false, false);
        acci = __builtin_amdgcn_wmma_f32_16x16x32_bf16(false, ai,  false, br, (short)0, acci, false, false);
    }
    if (col < BATCH) {
        #pragma unroll
        for (int v = 0; v < 8; ++v) {
            int o = t * 16 + v + 8 * kh;
            long idx = (((long)col * CW + o) * NROWS + r) * MODES + m;
            yfr[idx] = accr[v];
            yfi[idx] = acci[v];
        }
    }
}

// ---------------------------------------------------------------- inverse DFT along H
// Zw[bo, h, m] = sum_r Yf[bo, r, m] * e^{+2pi i kx_r h/264}
__global__ void fno_dft_h_inv(const float* __restrict__ yfr, const float* __restrict__ yfi,
                              const float* __restrict__ tab2,
                              float* __restrict__ zwr, float* __restrict__ zwi) {
    long idx = (long)blockIdx.x * blockDim.x + threadIdx.x;   // 8*64*264*32
    if (idx >= (long)BATCH * CW * HP * MODES) return;
    int m = (int)(idx & 31);
    long t = idx >> 5;
    int hh = (int)(t % HP);
    long bo = t / HP;
    const float* yr = yfr + bo * (long)NROWS * MODES + m;
    const float* yi = yfi + bo * (long)NROWS * MODES + m;
    float sr = 0.0f, si = 0.0f;
    #pragma unroll 4
    for (int r = 0; r < NROWS; ++r) {
        float c = tab2[r * HP + hh];
        float s = tab2[64 * HP + r * HP + hh];
        float a = yr[r * 32], b = yi[r * 32];
        sr += a * c - b * s;          // (a+ib)(c+is)
        si += a * s + b * c;
    }
    zwr[idx] = sr;
    zwi[idx] = si;
}

// ---------------------------------------------------------------- inverse DFT along W + add conv + GELU
// writes h (f32, in place) and the swizzled bf16 copy for the next layer's conv
__global__ void fno_inv_w_combine(const float* __restrict__ zwr, const float* __restrict__ zwi,
                                  const float* __restrict__ yconv, const float* __restrict__ tab1,
                                  float* __restrict__ hout, __bf16* __restrict__ hbf, int do_gelu) {
    long idx = (long)blockIdx.x * blockDim.x + threadIdx.x;   // 8*64*264*264
    if (idx >= (long)BATCH * CW * NPIX) return;
    long p = idx % NPIX;
    long bo = idx / NPIX;
    int  o = (int)(bo & 63);
    long b = bo >> 6;
    int  y = (int)(p % HP);
    long row = idx / HP;
    const float* zr = zwr + row * 32;
    const float* zi = zwi + row * 32;
    float s = zr[0];                                          // m=0 term (Im dropped, as irfft does)
    #pragma unroll 4
    for (int m = 1; m < MODES; ++m)
        s += 2.0f * (zr[m] * tab1[m * HP + y] - zi[m] * tab1[32 * HP + m * HP + y]);
    s *= (1.0f / (float)NPIX);                                // 1/(H*W) normalization
    float v = yconv[idx] + s;
    if (do_gelu) v = gelu_exact(v);
    hout[idx] = v;
    hbf[hbf_index(b, o, p)] = (__bf16)v;
}

// ---------------------------------------------------------------- crop
__global__ void fno_crop(const float* __restrict__ h, float* __restrict__ out) {
    long idx = (long)blockIdx.x * blockDim.x + threadIdx.x;   // 8*64*256*256
    if (idx >= (long)BATCH * CW * 256 * 256) return;
    int yy = (int)(idx & 255);
    long t = idx >> 8;
    int xx = (int)(t & 255);
    long bo = t >> 8;
    out[idx] = h[bo * (long)NPIX + (long)xx * HP + yy];
}

// ================================================================ launcher
extern "C" void kernel_launch(void* const* d_in, const int* in_sizes, int n_in,
                              void* d_out, int out_size, void* d_ws, size_t ws_size,
                              hipStream_t stream) {
    const float* x       = (const float*)d_in[0];
    const float* lift_w1 = (const float*)d_in[1];
    const float* lift_b1 = (const float*)d_in[2];
    const float* lift_w2 = (const float*)d_in[3];
    const float* lift_b2 = (const float*)d_in[4];
    const float* conv_w  = (const float*)d_in[5];
    const float* conv_b  = (const float*)d_in[6];
    const float* spw1    = (const float*)d_in[7];
    const float* spw2    = (const float*)d_in[8];
    float* out = (float*)d_out;

    char* ws = (char*)d_ws;
    const size_t szH   = (size_t)BATCH * CW * NPIX * 4;              // 142,737,408
    const size_t szHbf = (size_t)BATCH * NTILE * 2 * 32 * 16 * 2;    //  71,368,704 (bf16)
    const size_t szXw  = (size_t)BATCH * CW * HP * 32 * 4;           //  17,301,504
    const size_t szXfb = (size_t)BATCH * CW * NROWS * MODES * 2 * 2; //   4,194,304 (bf16)
    const size_t szYf  = (size_t)BATCH * CW * NROWS * MODES * 4;     //   4,194,304
    const size_t szWcb = (size_t)NROWS * MODES * 2 * CW * CW * 2;    //  33,554,432 (bf16)

    size_t off = 0;
    float*  hbuf = (float*)(ws + off);  off += szH;
    float*  ybuf = (float*)(ws + off);  off += szH;
    __bf16* hbf  = (__bf16*)(ws + off); off += szHbf;
    float*  xwr  = (float*)(ws + off);  off += szXw;     // reused as zwr
    float*  xwi  = (float*)(ws + off);  off += szXw;     // reused as zwi
    __bf16* xfb  = (__bf16*)(ws + off); off += szXfb;
    float*  yfr  = (float*)(ws + off);  off += szYf;
    float*  yfi  = (float*)(ws + off);  off += szYf;
    __bf16* wcb2 = (__bf16*)(ws + off); off += szWcb;
    float*  tab1 = (float*)(ws + off);  off += (size_t)2 * 32 * HP * 4;
    float*  tab2 = (float*)(ws + off);  off += (size_t)2 * 64 * HP * 4;
    (void)ws_size; (void)in_sizes; (void)n_in; (void)out_size;

    // twiddle tables (recomputed each call; no cross-call state)
    fno_init_tabs<<<(64 * HP + 255) / 256, 256, 0, stream>>>(tab1, tab2);

    // lift + pad (writes f32 h and swizzled bf16 copy)
    {
        long n = (long)BATCH * NPIX;
        fno_lift_pad<<<(unsigned)((n + 255) / 256), 256, 0, stream>>>(
            x, lift_w1, lift_b1, lift_w2, lift_b2, hbuf, hbf);
    }

    for (int k = 0; k < 4; ++k) {
        // stage spectral weights in bf16, A-fragment swizzled
        {
            long n = (long)NROWS * MODES * 2 * 4 * 2 * 32 * 16;
            fno_prep_weights<<<(unsigned)((n + 255) / 256), 256, 0, stream>>>(
                spw1, spw2, wcb2, k);
        }
        // 1x1 conv via WMMA (8 waves/block, weights in LDS)
        {
            dim3 grid((NTILE + 7) / 8, BATCH);
            fno_conv1x1_wmma<<<grid, 256, 0, stream>>>(
                hbf, conv_w + (long)k * CW * CW, conv_b + k * CW, ybuf);
        }
        // spectral path
        fno_dft_w_fwd<<<BATCH * CW * HP, 64, 0, stream>>>(hbuf, tab1, xwr, xwi);
        {
            long n = (long)BATCH * CW * NROWS * MODES;
            fno_dft_h_fwd<<<(unsigned)((n + 255) / 256), 256, 0, stream>>>(xwr, xwi, tab2, xfb);
        }
        {
            dim3 grid(NROWS * MODES, 4);
            fno_specmix_wmma<<<grid, 32, 0, stream>>>(xfb, wcb2, yfr, yfi);
        }
        {
            long n = (long)BATCH * CW * HP * MODES;
            fno_dft_h_inv<<<(unsigned)((n + 255) / 256), 256, 0, stream>>>(yfr, yfi, tab2, xwr, xwi);
        }
        // inverse W-DFT + conv add + activation; writes f32 h and bf16 swizzled h
        {
            long n = (long)BATCH * CW * NPIX;
            fno_inv_w_combine<<<(unsigned)((n + 255) / 256), 256, 0, stream>>>(
                xwr, xwi, ybuf, tab1, hbuf, hbf, (k < 3) ? 1 : 0);
        }
    }

    // crop to [8,64,256,256]
    {
        long n = (long)BATCH * CW * 256 * 256;
        fno_crop<<<(unsigned)((n + 255) / 256), 256, 0, stream>>>(hbuf, out);
    }
}
